// GINv2_38689065402516
// MI455X (gfx1250) — compile-verified, matrix operand order checked by reading, not run on hardware
//
#include <hip/hip_runtime.h>

#define DF 128  // feature dim (in = hid = out = 128)

typedef __attribute__((ext_vector_type(16))) __bf16 v16bf;
typedef __attribute__((ext_vector_type(8)))  float  v8f;

__device__ __forceinline__ __bf16 bf_hi(float f) { return (__bf16)f; }
__device__ __forceinline__ __bf16 bf_lo(float f, __bf16 h) { return (__bf16)(f - (float)h); }

// ---------------------------------------------------------------------------
// agg = cur  (float4 copy)
// ---------------------------------------------------------------------------
__global__ void k_copy4(const float4* __restrict__ in, float4* __restrict__ out, int n4) {
  int i = blockIdx.x * blockDim.x + threadIdx.x;
  if (i < n4) out[i] = in[i];
}

// ---------------------------------------------------------------------------
// agg[dst[e]] += cur[src[e]]  — one wave32 per edge, float4 per lane,
// non-returning relaxed agent-scope f32 atomics (global_atomic_add_f32).
// ---------------------------------------------------------------------------
__global__ void k_scatter(const float* __restrict__ h, const int* __restrict__ src,
                          const int* __restrict__ dst, float* __restrict__ agg, int n_edges) {
  int e    = (blockIdx.x * blockDim.x + threadIdx.x) >> 5;
  int lane = threadIdx.x & 31;
  if (e >= n_edges) return;
  int s = src[e], d = dst[e];
  const float4 v = *(const float4*)(h + (size_t)s * DF + lane * 4);
  float* o = agg + (size_t)d * DF + lane * 4;
  __hip_atomic_fetch_add(o + 0, v.x, __ATOMIC_RELAXED, __HIP_MEMORY_SCOPE_AGENT);
  __hip_atomic_fetch_add(o + 1, v.y, __ATOMIC_RELAXED, __HIP_MEMORY_SCOPE_AGENT);
  __hip_atomic_fetch_add(o + 2, v.z, __ATOMIC_RELAXED, __HIP_MEMORY_SCOPE_AGENT);
  __hip_atomic_fetch_add(o + 3, v.w, __ATOMIC_RELAXED, __HIP_MEMORY_SCOPE_AGENT);
}

// ---------------------------------------------------------------------------
// Convert a 128x128 f32 weight (row-major [k][n]) into LDS as hi/lo bf16,
// pre-swizzled into the exact WMMA B-fragment layout so each (kt,nt) fragment
// is 16 contiguous bf16 per lane:
//   frag base = ((nt*4 + kt)*32 + lane)*16
//   elem j<8  -> K = kt*32 + 8*(lane>=16) + j
//   elem j>=8 -> K = kt*32 + 16 + 8*(lane>=16) + (j-8),  N = nt*16 + (lane&15)
// ---------------------------------------------------------------------------
__device__ __forceinline__ void load_w_swizzled(const float* __restrict__ W,
                                                __bf16* sWh, __bf16* sWl, int tid) {
  const int n = tid;  // 128 threads: one column each
  for (int k = 0; k < DF; ++k) {
    float w   = W[k * DF + n];
    int   kin = k & 31;
    int   pos = (((n >> 4) * 4 + (k >> 5)) * 32 + ((n & 15) + 16 * ((kin >> 3) & 1))) * 16
                + (kin & 7) + 8 * ((kin >> 4) & 1);
    __bf16 h = bf_hi(w);
    sWh[pos] = h;
    sWl[pos] = bf_lo(w, h);
  }
}

// ---------------------------------------------------------------------------
// Fused MLP: out = [LN+ReLU]( relu(agg@W1+b1) @ W2 + b2 )
// 4 waves/WG, each wave: 16 rows x 128 cols.  bf16x3 split WMMA, f32 accum.
// Dynamic LDS: sWh(32K) sWl(32K) sAh(16K) sAl(16K) = 96KB.
// ---------------------------------------------------------------------------
__global__ __launch_bounds__(128) void k_mlp(
    const float* __restrict__ agg,
    const float* __restrict__ w1, const float* __restrict__ b1,
    const float* __restrict__ w2, const float* __restrict__ b2,
    const float* __restrict__ gamma, const float* __restrict__ beta,
    float* __restrict__ out, int n_nodes, int do_ln) {
  extern __shared__ char smem[];
  __bf16* sWh = (__bf16*)(smem);
  __bf16* sWl = (__bf16*)(smem + 32768);
  __bf16* sAh = (__bf16*)(smem + 65536);
  __bf16* sAl = (__bf16*)(smem + 81920);

  const int  tid    = threadIdx.x;
  const int  wv     = tid >> 5;
  const int  lane   = tid & 31;
  const int  nl     = lane & 15;
  const int  half   = lane >> 4;          // K-half selector of this lane
  const int  m0     = (blockIdx.x * 4 + wv) * 16;
  const bool active = (m0 < n_nodes);     // wave-uniform

  // ---- phase 1: W1 into LDS ----
  load_w_swizzled(w1, sWh, sWl, tid);
  __syncthreads();

  const v8f vzero = {0.f, 0.f, 0.f, 0.f, 0.f, 0.f, 0.f, 0.f};
  v8f acc[8];
#pragma unroll
  for (int nt = 0; nt < 8; ++nt) acc[nt] = vzero;

  if (active) {
    int row = m0 + nl;
    if (row >= n_nodes) row = n_nodes - 1;  // clamp (n_nodes%16==0 in practice)
    const float* base = agg + (size_t)row * DF;
#pragma unroll
    for (int kt = 0; kt < 4; ++kt) {
      // Build A fragment from global f32, split to hi/lo bf16.
      const float* p = base + kt * 32 + half * 8;
      float va[16];
      *(float4*)(va + 0)  = *(const float4*)(p + 0);
      *(float4*)(va + 4)  = *(const float4*)(p + 4);
      *(float4*)(va + 8)  = *(const float4*)(p + 16);
      *(float4*)(va + 12) = *(const float4*)(p + 20);
      v16bf ah, al;
#pragma unroll
      for (int j = 0; j < 16; ++j) { __bf16 h = bf_hi(va[j]); ah[j] = h; al[j] = bf_lo(va[j], h); }
#pragma unroll
      for (int nt = 0; nt < 8; ++nt) {
        const v16bf bh = *(const v16bf*)(sWh + ((nt * 4 + kt) * 32 + lane) * 16);
        const v16bf bl = *(const v16bf*)(sWl + ((nt * 4 + kt) * 32 + lane) * 16);
        acc[nt] = __builtin_amdgcn_wmma_f32_16x16x32_bf16(false, ah, false, bh, (short)0, acc[nt], false, false);
        acc[nt] = __builtin_amdgcn_wmma_f32_16x16x32_bf16(false, ah, false, bl, (short)0, acc[nt], false, false);
        acc[nt] = __builtin_amdgcn_wmma_f32_16x16x32_bf16(false, al, false, bh, (short)0, acc[nt], false, false);
      }
    }
    // bias + ReLU, then park h1 into LDS already in A-fragment hi/lo layout.
    __bf16* mAh = sAh + wv * 2048;
    __bf16* mAl = sAl + wv * 2048;
#pragma unroll
    for (int nt = 0; nt < 8; ++nt) {
      float bb    = b1[nt * 16 + nl];
      int   k     = nt * 16 + nl;         // h1 column -> GEMM2 K index
      int   kt2   = k >> 5;
      int   elem  = (k & 7) + 8 * ((k >> 4) & 1);
      int   laneK = 16 * ((k >> 3) & 1);
#pragma unroll
      for (int r = 0; r < 8; ++r) {
        float v = fmaxf(acc[nt][r] + bb, 0.0f);
        int   m   = half * 8 + r;         // D-layout row
        int   pos = (kt2 * 32 + (m + laneK)) * 16 + elem;
        __bf16 h  = bf_hi(v);
        mAh[pos] = h;
        mAl[pos] = bf_lo(v, h);
      }
    }
  }
  __syncthreads();

  // ---- phase 2: W2 overwrites weight LDS ----
  load_w_swizzled(w2, sWh, sWl, tid);
  __syncthreads();

#pragma unroll
  for (int nt = 0; nt < 8; ++nt) acc[nt] = vzero;
  if (!active) return;  // no more barriers below

  const __bf16* mAh = sAh + wv * 2048;
  const __bf16* mAl = sAl + wv * 2048;
#pragma unroll
  for (int kt = 0; kt < 4; ++kt) {
    const v16bf ah = *(const v16bf*)(mAh + (kt * 32 + lane) * 16);
    const v16bf al = *(const v16bf*)(mAl + (kt * 32 + lane) * 16);
#pragma unroll
    for (int nt = 0; nt < 8; ++nt) {
      const v16bf bh = *(const v16bf*)(sWh + ((nt * 4 + kt) * 32 + lane) * 16);
      const v16bf bl = *(const v16bf*)(sWl + ((nt * 4 + kt) * 32 + lane) * 16);
      acc[nt] = __builtin_amdgcn_wmma_f32_16x16x32_bf16(false, ah, false, bh, (short)0, acc[nt], false, false);
      acc[nt] = __builtin_amdgcn_wmma_f32_16x16x32_bf16(false, ah, false, bl, (short)0, acc[nt], false, false);
      acc[nt] = __builtin_amdgcn_wmma_f32_16x16x32_bf16(false, al, false, bh, (short)0, acc[nt], false, false);
    }
  }

  float bv[8], gv[8], btv[8];
#pragma unroll
  for (int nt = 0; nt < 8; ++nt) {
    bv[nt] = b2[nt * 16 + nl];
    if (do_ln) { gv[nt] = gamma[nt * 16 + nl]; btv[nt] = beta[nt * 16 + nl]; }
  }

  if (do_ln) {
    // LayerNorm per row: row values live in one 16-lane half (D layout),
    // so a width-16 xor-butterfly gives full row sums (rows r and r+8 at once).
#pragma unroll
    for (int r = 0; r < 8; ++r) {
      float vr[8], s = 0.f, ss = 0.f;
#pragma unroll
      for (int nt = 0; nt < 8; ++nt) {
        float v = acc[nt][r] + bv[nt];
        vr[nt] = v; s += v; ss += v * v;
      }
#pragma unroll
      for (int off = 1; off < 16; off <<= 1) {
        s  += __shfl_xor(s,  off, 32);
        ss += __shfl_xor(ss, off, 32);
      }
      float mu  = s * (1.0f / DF);
      float var = ss * (1.0f / DF) - mu * mu;
      float rs  = rsqrtf(var + 1e-5f);
      int rowg = m0 + half * 8 + r;
      if (rowg < n_nodes) {
        float* op = out + (size_t)rowg * DF + nl;
#pragma unroll
        for (int nt = 0; nt < 8; ++nt) {
          float v = (vr[nt] - mu) * rs * gv[nt] + btv[nt];
          op[nt * 16] = fmaxf(v, 0.0f);  // post-LN ReLU
        }
      }
    }
  } else {
    // final block: bias only, no norm / activation
#pragma unroll
    for (int r = 0; r < 8; ++r) {
      int rowg = m0 + half * 8 + r;
      if (rowg < n_nodes) {
        float* op = out + (size_t)rowg * DF + nl;
#pragma unroll
        for (int nt = 0; nt < 8; ++nt) op[nt * 16] = acc[nt][r] + bv[nt];
      }
    }
  }
}

// ---------------------------------------------------------------------------
extern "C" void kernel_launch(void* const* d_in, const int* in_sizes, int n_in,
                              void* d_out, int out_size, void* d_ws, size_t ws_size,
                              hipStream_t stream) {
  const float* x  = (const float*)d_in[0];
  const int*   ei = (const int*)d_in[1];
  const int n_nodes = in_sizes[0] / DF;
  const int n_edges = in_sizes[1] / 2;
  const int* src = ei;
  const int* dst = ei + n_edges;

  const float *w1[4], *b1[4], *w2[4], *b2[4];
  for (int i = 0; i < 4; ++i) {
    w1[i] = (const float*)d_in[2 + 4 * i + 0];
    b1[i] = (const float*)d_in[2 + 4 * i + 1];
    w2[i] = (const float*)d_in[2 + 4 * i + 2];
    b2[i] = (const float*)d_in[2 + 4 * i + 3];
  }
  const float *g[3], *bt[3];
  for (int i = 0; i < 3; ++i) {
    g[i]  = (const float*)d_in[18 + 2 * i];
    bt[i] = (const float*)d_in[19 + 2 * i];
  }

  float* agg = (float*)d_ws;    // 25.6 MB aggregation buffer
  float* h   = (float*)d_out;   // d_out doubles as the feature ping buffer

  const int    n4     = n_nodes * (DF / 4);
  const int    cgrid  = (n4 + 255) / 256;
  const int    sgrid  = (n_edges * 32 + 255) / 256;  // one wave32 per edge
  const int    mgrid  = (n_nodes + 63) / 64;         // 4 waves x 16 rows per WG
  const size_t ldsB   = 98304;                       // 96 KB dynamic LDS

  const float* cur = x;
  for (int blk = 0; blk < 4; ++blk) {
    k_copy4<<<cgrid, 256, 0, stream>>>((const float4*)cur, (float4*)agg, n4);
    k_scatter<<<sgrid, 256, 0, stream>>>(cur, src, dst, agg, n_edges);
    const int do_ln = (blk < 3) ? 1 : 0;
    k_mlp<<<mgrid, 128, ldsB, stream>>>(agg, w1[blk], b1[blk], w2[blk], b2[blk],
                                        do_ln ? g[blk] : nullptr,
                                        do_ln ? bt[blk] : nullptr,
                                        h, n_nodes, do_ln);
    cur = h;
  }
}
